// SimpleConv2D_14388140442126
// MI455X (gfx1250) — matrix operand order repaired and checked by reading, not compiled, not run on hardware
//
#include <hip/hip_runtime.h>
#include <hip/hip_bf16.h>

typedef __attribute__((ext_vector_type(8)))  float  v8f;
typedef __attribute__((ext_vector_type(16))) __bf16 v16bf;
typedef __attribute__((ext_vector_type(8)))  __bf16 v8bf;

// Problem constants: x(32,64,112,112) f32, w(128,64,3,3) f32, bias(128), VALID, stride 1.
#define NB    32
#define CIN   64
#define HIN   112
#define WIN   112
#define XHW   (HIN*WIN)           // 12544
#define COUT  128
#define HOUT  110
#define WOUT  110
#define KTOT  576                 // 9 * CIN, khw-major: k = khw*64 + c
#define KT    32                  // K per WMMA step (bf16)
#define NSTEP (KTOT/KT)           // 18 ; khw = s>>1, c0 = (s&1)*32
#define BN    128                 // pixels per block
#define PIX_PER_IMG (HOUT*WOUT)   // 12100

// ---- Pre-pass: reorder weight (cout, c, kh, kw) f32 -> (cout, khw, c) bf16 in d_ws ----
__global__ __launch_bounds__(256)
void prep_weight_bf16(const float* __restrict__ w, __bf16* __restrict__ Ap) {
    const int idx  = blockIdx.x * 256 + threadIdx.x;   // < COUT*KTOT = 73728
    const int cout = idx / KTOT;
    const int r    = idx - cout * KTOT;                // destination k (khw-major)
    const int khw  = r >> 6;                           // 0..8
    const int c    = r & 63;
    Ap[idx] = (__bf16)w[cout * KTOT + c * 9 + khw];
}

__global__ __launch_bounds__(256)
void conv2d_implicit_gemm_wmma(const float* __restrict__ x,
                               const __bf16* __restrict__ Ap,   // reordered bf16 weight
                               const float* __restrict__ bias,
                               float* __restrict__ out)
{
    __shared__ __bf16 sB[BN * KT];     // B tile transposed: [pixel 0..127][k 0..31], 8 KB

    const int t      = threadIdx.x;
    const int lane   = t & 31;
    const int laneLo = lane & 15;
    const int half   = lane >> 4;
    const int wave   = t >> 5;         // 0..7
    const int wm     = wave & 3;       // 4 waves * 32 rows = 128 couts
    const int wn     = wave >> 2;      // 2 waves * 64 cols = 128 pixels

    const int p0 = blockIdx.x * BN;

    // Accumulators: wave tile 32x64 = 2x4 subtiles of 16x16 f32
    v8f acc[2][4];
    #pragma unroll
    for (int m = 0; m < 2; ++m)
        #pragma unroll
        for (int n = 0; n < 4; ++n)
            acc[m][n] = (v8f){0.f,0.f,0.f,0.f,0.f,0.f,0.f,0.f};

    // ---- B staging thread mapping: one pixel-row half (16 k values) per thread ----
    const int jpix  = t >> 1;                // pixel slot 0..127
    const int kb    = (t & 1) * 16;          // k_local base 0 / 16
    const int pB    = p0 + jpix;
    const int nimgB = pB / PIX_PER_IMG;
    const int remB  = pB - nimgB * PIX_PER_IMG;
    const int ohB   = remB / WOUT;
    const int owB   = remB - ohB * WOUT;
    const float* xPix = x + (long)nimgB * CIN * XHW + ohB * WIN + owB;  // + kh*WIN + kw + c*XHW

    __bf16* sBrow = &sB[jpix * KT + kb];

    for (int s = 0; s < NSTEP; ++s) {
        const int khw = s >> 1;              // 0..8, uniform per step
        const int c0  = (s & 1) << 5;        // 0 / 32
        const int kh  = khw / 3;
        const int kw  = khw - kh * 3;

        // ---- Stage B: im2col gather, linear in channel (no per-element division) ----
        const float* base2 = xPix + kh * WIN + kw + (c0 + kb) * XHW;
        float fv[16];
        #pragma unroll
        for (int q = 0; q < 16; ++q)
            fv[q] = base2[q * XHW];
        __bf16 bvv[16];
        #pragma unroll
        for (int q = 0; q < 16; ++q)
            bvv[q] = (__bf16)fv[q];
        *reinterpret_cast<v8bf*>(sBrow)     = *reinterpret_cast<v8bf*>(&bvv[0]);
        *reinterpret_cast<v8bf*>(sBrow + 8) = *reinterpret_cast<v8bf*>(&bvv[8]);

        // Prefetch next step's gather window while we compute
        if (s + 1 < NSTEP) {
            const int khw2 = (s + 1) >> 1;
            const int c02  = ((s + 1) & 1) << 5;
            __builtin_prefetch(xPix + (khw2 / 3) * WIN + (khw2 - (khw2 / 3) * 3)
                                    + (c02 + kb) * XHW, 0, 0);
        }

        // ---- A fragments straight from global: contiguous 16B chunks, contiguous k order (k = 16h + s), same as B ----
        v16bf aF[2];
        #pragma unroll
        for (int m = 0; m < 2; ++m) {
            const __bf16* ar = Ap + (long)(wm * 32 + m * 16 + laneLo) * KTOT + khw * 64 + c0;
            v8bf a0 = *reinterpret_cast<const v8bf*>(ar + half * 16);       // k: h*16 .. +7
            v8bf a1 = *reinterpret_cast<const v8bf*>(ar + half * 16 + 8);   // k: h*16+8 .. +7
            aF[m] = __builtin_shufflevector(a0, a1, 0,1,2,3,4,5,6,7,8,9,10,11,12,13,14,15);
        }

        __syncthreads();

        // ---- B fragments from LDS ----
        v16bf bF[4];
        #pragma unroll
        for (int n = 0; n < 4; ++n) {
            const __bf16* base = &sB[(wn * 64 + n * 16 + laneLo) * KT] + half * 16; // k: h*16..+15
            v8bf b0 = *reinterpret_cast<const v8bf*>(base);
            v8bf b1 = *reinterpret_cast<const v8bf*>(base + 8);
            bF[n] = __builtin_shufflevector(b0, b1, 0,1,2,3,4,5,6,7,8,9,10,11,12,13,14,15);
        }

        // ---- 2x4 WMMA: D = A x B + C ----
        #pragma unroll
        for (int m = 0; m < 2; ++m)
            #pragma unroll
            for (int n = 0; n < 4; ++n)
                acc[m][n] = __builtin_amdgcn_wmma_f32_16x16x32_bf16(
                    false, aF[m], false, bF[n], (short)0, acc[m][n], false, false);

        __syncthreads();
    }

    // ---- Epilogue: bias add + scatter to NCHW fp32 output ----
    #pragma unroll
    for (int m = 0; m < 2; ++m) {
        const int cbase = wm * 32 + m * 16 + half * 8;   // cout of accumulator VGPR r = cbase + r
        float bv[8];
        #pragma unroll
        for (int r = 0; r < 8; ++r) bv[r] = bias[cbase + r];

        #pragma unroll
        for (int n = 0; n < 4; ++n) {
            const int p    = p0 + wn * 64 + n * 16 + laneLo;
            const int nimg = p / PIX_PER_IMG;
            const int rem  = p - nimg * PIX_PER_IMG;
            const int oh   = rem / WOUT;
            const int ow   = rem - oh * WOUT;
            float* obase = out + (((long)nimg * COUT + cbase) * HOUT + oh) * WOUT + ow;
            #pragma unroll
            for (int r = 0; r < 8; ++r)
                obase[(long)r * (HOUT * WOUT)] = acc[m][n][r] + bv[r];
        }
    }
}

extern "C" void kernel_launch(void* const* d_in, const int* in_sizes, int n_in,
                              void* d_out, int out_size, void* d_ws, size_t ws_size,
                              hipStream_t stream) {
    const float* x    = (const float*)d_in[0];
    const float* wgt  = (const float*)d_in[1];
    const float* bias = (const float*)d_in[2];
    float* out        = (float*)d_out;
    __bf16* Ap        = (__bf16*)d_ws;   // needs COUT*KTOT*2 = 147456 bytes

    // 1) Reorder weight to bf16 (cout, khw, c) — makes A fragments contiguous and
    //    removes all division from the hot im2col gather.
    prep_weight_bf16<<<dim3((COUT * KTOT) / 256), dim3(256), 0, stream>>>(wgt, Ap);

    // 2) Implicit-GEMM conv: M=128 (cout), N=387200 pixels, K=576; 128x128 per block.
    const int nblocks = (NB * HOUT * WOUT) / BN;   // 3025, exact
    conv2d_implicit_gemm_wmma<<<dim3(nblocks), dim3(256), 0, stream>>>(x, Ap, bias, out);
}
